// GlycanMPNN_34995393527946
// MI455X (gfx1250) — compile-verified, matrix operand order checked by reading, not run on hardware
//
#include <hip/hip_runtime.h>
#include <cstddef>
#include <cstdint>

// Problem constants (match reference)
#define HDIM   128
#define NB     512      // graphs
#define NNODE  8192     // nodes
#define NEDGE  16384    // edges
#define EDIM   32
#define ECHUNK 2048     // edges per transform chunk (64 MiB f16 chunk -> lives in L2)
#define NLAYER 3
#define NS2S   3

typedef _Float16 v8h  __attribute__((ext_vector_type(8)));
typedef _Float16 v16h __attribute__((ext_vector_type(16)));
typedef float    v8f  __attribute__((ext_vector_type(8)));

// ---------------------------------------------------------------------------
// WMMA fragment loader. Row-major source with leading dim `ld` (in halves).
// 16-bit A-matrix 16x32 layout (wave32):
//   lanes 0-15  (M=lane):    halves 0..7 = K 0..7,  halves 8..15 = K 16..23
//   lanes 16-31 (M=lane-16): halves 0..7 = K 8..15, halves 8..15 = K 24..31
// B (32x16) is symmetric with lane = column, so loading from the
// pre-transposed Bt [N,K] uses the identical pattern.
// ---------------------------------------------------------------------------
__device__ __forceinline__ v16h load_frag(const _Float16* __restrict__ p0,
                                          int ld, int lane) {
  const _Float16* p = p0 + (size_t)(lane & 15) * ld + ((lane >> 4) << 3);
  v8h lo = *(const v8h*)(p);
  v8h hi = *(const v8h*)(p + 16);
  return __builtin_shufflevector(lo, hi, 0, 1, 2, 3, 4, 5, 6, 7,
                                 8, 9, 10, 11, 12, 13, 14, 15);
}

// ---------------------------------------------------------------------------
// WMMA GEMM, 4x register-blocked along N: each wave computes a 16x64 strip
// (4 accumulators sharing one A fragment per K-step -> 4x less A traffic).
// All 5 fragment loads of a K-step are issued BEFORE the 4 WMMAs so the
// in-order LOADcnt lets each WMMA overlap with the remaining loads.
// C[M,N] = A[M,K] * B[K,N] + bias[N]  (optional relu)
// A: row-major f16 [M,K].  Bt: B transposed, row-major f16 [N,K].
// 256 threads = 8 waves per block; N must be a multiple of 64.
// ---------------------------------------------------------------------------
template <bool OUTH, bool RELU>
__global__ __launch_bounds__(256) void wmma_gemm(
    const _Float16* __restrict__ A, const _Float16* __restrict__ Bt,
    const float* __restrict__ bias, void* __restrict__ Cv,
    int M, int N, int K) {
  const int lane = threadIdx.x & 31;
  const int wave = threadIdx.x >> 5;
  const int groupsN = N >> 6;                      // 64-column groups
  const long total = (long)(M >> 4) * groupsN;
  long tile = (long)blockIdx.x * 8 + wave;
  if (tile >= total) return;                       // wave-uniform, EXEC stays all-ones
  const int tm = (int)(tile / groupsN);
  const int tn = (int)(tile % groupsN);
  const _Float16* Ab = A + (size_t)tm * 16 * K;
  const _Float16* Bb = Bt + (size_t)tn * 64 * K;

  v8f acc[4];
#pragma unroll
  for (int j = 0; j < 4; ++j) {
    const float bv = bias ? bias[tn * 64 + j * 16 + (lane & 15)] : 0.0f;
#pragma unroll
    for (int i = 0; i < 8; ++i) acc[j][i] = bv;    // fold bias into accumulator init
  }

  for (int k0 = 0; k0 < K; k0 += 32) {
    // issue every load of this K-step first (one clause), WMMAs after
    v16h a = load_frag(Ab + k0, K, lane);
    v16h b0 = load_frag(Bb + (size_t)0 * 16 * K + k0, K, lane);
    v16h b1 = load_frag(Bb + (size_t)1 * 16 * K + k0, K, lane);
    v16h b2 = load_frag(Bb + (size_t)2 * 16 * K + k0, K, lane);
    v16h b3 = load_frag(Bb + (size_t)3 * 16 * K + k0, K, lane);
    acc[0] = __builtin_amdgcn_wmma_f32_16x16x32_f16(
        false, a, false, b0, (short)0, acc[0], false, false);
    acc[1] = __builtin_amdgcn_wmma_f32_16x16x32_f16(
        false, a, false, b1, (short)0, acc[1], false, false);
    acc[2] = __builtin_amdgcn_wmma_f32_16x16x32_f16(
        false, a, false, b2, (short)0, acc[2], false, false);
    acc[3] = __builtin_amdgcn_wmma_f32_16x16x32_f16(
        false, a, false, b3, (short)0, acc[3], false, false);
  }

  const int row0 = tm * 16 + ((lane >> 4) << 3);
#pragma unroll
  for (int j = 0; j < 4; ++j) {
    const int col = tn * 64 + j * 16 + (lane & 15);
    if (OUTH) {
      _Float16* C = (_Float16*)Cv;
#pragma unroll
      for (int r = 0; r < 8; ++r) {
        float v = acc[j][r];
        if (RELU) v = fmaxf(v, 0.0f);
        C[(size_t)(row0 + r) * N + col] = (_Float16)v;
      }
    } else {
      float* C = (float*)Cv;
#pragma unroll
      for (int r = 0; r < 8; ++r) {
        float v = acc[j][r];
        if (RELU) v = fmaxf(v, 0.0f);
        C[(size_t)(row0 + r) * N + col] = v;
      }
    }
  }
}

// ---------------------------------------------------------------------------
// Small elementwise / data-movement kernels
// ---------------------------------------------------------------------------
__global__ void cvt_f32_f16(const float* __restrict__ s, _Float16* __restrict__ d, int n) {
  int i = blockIdx.x * blockDim.x + threadIdx.x;
  if (i < n) d[i] = (_Float16)s[i];
}

// src [R,C] f32 -> dst [C,R] f16 (transpose+convert; i indexes dst)
__global__ void trans_cvt(const float* __restrict__ s, _Float16* __restrict__ d,
                          int R, int C) {
  int i = blockIdx.x * blockDim.x + threadIdx.x;
  if (i >= R * C) return;
  int c = i / R, r = i - c * R;
  d[i] = (_Float16)s[(size_t)r * C + c];
}

__global__ void zero_f32(float* __restrict__ p, int n) {
  int i = blockIdx.x * blockDim.x + threadIdx.x;
  if (i < n) p[i] = 0.0f;
}

__global__ void copy_f32(const float* __restrict__ s, float* __restrict__ d, int n) {
  int i = blockIdx.x * blockDim.x + threadIdx.x;
  if (i < n) d[i] = s[i];
}

__global__ void gather_emb(const int* __restrict__ ut, const float* __restrict__ emb,
                           float* __restrict__ hf, int total) {
  int i = blockIdx.x * blockDim.x + threadIdx.x;
  if (i >= total) return;
  int n = i >> 7, h = i & (HDIM - 1);
  hf[i] = emb[(size_t)ut[n] * HDIM + h];
}

// x = relu(update) -> f16, h(f32) -> f16 (GEMM inputs for the GRU gates)
__global__ void relu_cvt(const float* __restrict__ upd, const float* __restrict__ hf,
                         _Float16* __restrict__ x16, _Float16* __restrict__ h16,
                         int total) {
  int i = blockIdx.x * blockDim.x + threadIdx.x;
  if (i >= total) return;
  x16[i] = (_Float16)fmaxf(upd[i], 0.0f);
  h16[i] = (_Float16)hf[i];
}

__device__ __forceinline__ float sigm(float x) {
  return 1.0f / (1.0f + __expf(-x));
}

// GRU gate order: reset, update, new (torch convention)
__global__ void gru_elem(const float* __restrict__ gi, const float* __restrict__ gh,
                         float* __restrict__ hx, int total) {
  int i = blockIdx.x * blockDim.x + threadIdx.x;
  if (i >= total) return;
  int n = i >> 7, h = i & (HDIM - 1);
  const float* a = gi + (size_t)n * 3 * HDIM;
  const float* b = gh + (size_t)n * 3 * HDIM;
  float r  = sigm(a[h] + b[h]);
  float z  = sigm(a[HDIM + h] + b[HDIM + h]);
  float nn = tanhf(a[2 * HDIM + h] + r * b[2 * HDIM + h]);
  hx[i] = (1.0f - z) * nn + z * hx[i];
}

// LSTM gate order: input, forget, cell, output
__global__ void lstm_elem(const float* __restrict__ g1, const float* __restrict__ g2,
                          float* __restrict__ h, float* __restrict__ c, int total) {
  int i = blockIdx.x * blockDim.x + threadIdx.x;
  if (i >= total) return;
  int b = i >> 7, j = i & (HDIM - 1);
  const float* p = g1 + (size_t)b * 4 * HDIM;
  const float* q = g2 + (size_t)b * 4 * HDIM;
  float gi = p[j] + q[j];
  float gf = p[HDIM + j] + q[HDIM + j];
  float gg = p[2 * HDIM + j] + q[2 * HDIM + j];
  float go = p[3 * HDIM + j] + q[3 * HDIM + j];
  float cn = sigm(gf) * c[i] + sigm(gi) * tanhf(gg);
  c[i] = cn;
  h[i] = sigm(go) * tanhf(cn);
}

// ---------------------------------------------------------------------------
// Per-edge matvec + scatter: msg[e] = T[e](128x128, f16) @ x[node_in[e]],
// atomically accumulated into update[node_out[e]]. One 128-thread block/edge.
// The gathered source-node vector is staged into LDS with the CDNA5 async
// global->LDS path (GLOBAL_LOAD_ASYNC_TO_LDS_B32, tracked by ASYNCcnt):
// each lane supplies its own global address + LDS offset.
// ---------------------------------------------------------------------------
__global__ __launch_bounds__(128) void edge_msg(
    const _Float16* __restrict__ Tchunk, const float* __restrict__ hfeat,
    const int* __restrict__ node_in, const int* __restrict__ node_out,
    float* __restrict__ update, int e0) {
  const int e = e0 + blockIdx.x;
  const int tid = threadIdx.x;
  __shared__ float xs[HDIM];

  const float* gaddr = hfeat + (size_t)node_in[e] * HDIM + tid;
  // Flat LDS addresses carry the LDS byte offset in addr[31:0] (ISA aperture
  // rules), so truncating the generic pointer yields the async VDST operand.
  unsigned lds_off = (unsigned)(uintptr_t)(&xs[tid]);
  asm volatile("global_load_async_to_lds_b32 %0, %1, off"
               :: "v"(lds_off), "v"(gaddr) : "memory");
  asm volatile("s_wait_asynccnt 0" ::: "memory");
  __syncthreads();

  const _Float16* row = Tchunk + ((size_t)blockIdx.x * HDIM + tid) * HDIM;
  float acc = 0.0f;
#pragma unroll
  for (int j = 0; j < HDIM; j += 8) {
    v8h t = *(const v8h*)(row + j);
#pragma unroll
    for (int k = 0; k < 8; ++k) acc += (float)t[k] * xs[j + k];
  }
  atomicAdd(&update[(size_t)node_out[e] * HDIM + tid], acc);
}

// node2graph is sorted -> compute segment start offsets: start[b] = first node of b
__global__ void seg_starts(const int* __restrict__ seg, int* __restrict__ start,
                           int n, int b) {
  int i = blockIdx.x * blockDim.x + threadIdx.x;
  if (i >= n) return;
  int s = seg[i];
  int sp = (i == 0) ? -1 : seg[i - 1];
  for (int g = sp + 1; g <= s; ++g) start[g] = i;
  if (i == n - 1)
    for (int g = s + 1; g <= b; ++g) start[g] = n;
}

// Set2Set attention + pooling, one block (128 threads) per graph.
// Two-pass segment softmax (max, then exp/sum/pool); writes q_star directly.
__global__ __launch_bounds__(128) void s2s_attn(
    const float* __restrict__ feat, const float* __restrict__ h,
    const int* __restrict__ start, float* __restrict__ qstar) {
  const int b = blockIdx.x, tid = threadIdx.x;
  __shared__ float hv[HDIM];
  __shared__ float red[HDIM];
  hv[tid] = h[(size_t)b * HDIM + tid];
  __syncthreads();
  const int s = start[b], e = start[b + 1];

  float m = -1e30f;
  for (int n = s; n < e; ++n) {
    red[tid] = feat[(size_t)n * HDIM + tid] * hv[tid];
    __syncthreads();
    for (int off = 64; off > 0; off >>= 1) {
      if (tid < off) red[tid] += red[tid + off];
      __syncthreads();
    }
    m = fmaxf(m, red[0]);
    __syncthreads();
  }

  float pooled = 0.0f, denom = 0.0f;
  for (int n = s; n < e; ++n) {
    float f = feat[(size_t)n * HDIM + tid];
    red[tid] = f * hv[tid];
    __syncthreads();
    for (int off = 64; off > 0; off >>= 1) {
      if (tid < off) red[tid] += red[tid + off];
      __syncthreads();
    }
    float d = red[0];
    __syncthreads();
    float ex = __expf(d - m);
    denom += ex;
    pooled += ex * f;
  }
  qstar[(size_t)b * 2 * HDIM + tid] = hv[tid];
  qstar[(size_t)b * 2 * HDIM + HDIM + tid] = (denom > 0.0f) ? pooled / denom : 0.0f;
}

// ---------------------------------------------------------------------------
static inline int cdiv(long a, long b) { return (int)((a + b - 1) / b); }

extern "C" void kernel_launch(void* const* d_in, const int* in_sizes, int n_in,
                              void* d_out, int out_size, void* d_ws, size_t ws_size,
                              hipStream_t stream) {
  (void)in_sizes; (void)n_in; (void)out_size; (void)ws_size;

  const int*   unit_type   = (const int*)d_in[0];
  const int*   node_in     = (const int*)d_in[1];
  const int*   node_out    = (const int*)d_in[2];
  const int*   node2graph  = (const int*)d_in[3];
  const float* edge_feat   = (const float*)d_in[4];
  const float* embedding   = (const float*)d_in[5];
  const float* mlp_w1      = (const float*)d_in[6];
  const float* mlp_b1      = (const float*)d_in[7];
  const float* mlp_w2      = (const float*)d_in[8];
  const float* mlp_b2      = (const float*)d_in[9];
  const float* gru_w_ih    = (const float*)d_in[10];
  const float* gru_w_hh    = (const float*)d_in[11];
  const float* gru_b_ih    = (const float*)d_in[12];
  const float* gru_b_hh    = (const float*)d_in[13];
  const float* lstm_w_ih   = (const float*)d_in[14];
  const float* lstm_w_hh   = (const float*)d_in[15];
  const float* lstm_b_ih   = (const float*)d_in[16];
  const float* lstm_b_hh   = (const float*)d_in[17];
  float* out = (float*)d_out;

  // ---- workspace carve-up (all 256B aligned; total ~115 MiB) ----
  size_t off = 0;
  char* base = (char*)d_ws;
  auto alloc = [&](size_t bytes) -> void* {
    void* p = base + off;
    off += (bytes + 255) & ~(size_t)255;
    return p;
  };
  _Float16* Tchunk  = (_Float16*)alloc((size_t)ECHUNK * HDIM * HDIM * 2);  // 64 MiB
  _Float16* hmid16  = (_Float16*)alloc((size_t)NEDGE * HDIM * 2);          // 4 MiB
  _Float16* edge16  = (_Float16*)alloc((size_t)NEDGE * EDIM * 2);          // 1 MiB
  _Float16* W1t     = (_Float16*)alloc((size_t)HDIM * EDIM * 2);           // Bt [128,32]
  _Float16* W2t     = (_Float16*)alloc((size_t)HDIM * HDIM * HDIM * 2);    // Bt [16384,128]
  _Float16* gruIH   = (_Float16*)alloc((size_t)3 * HDIM * HDIM * 2);       // Bt [384,128]
  _Float16* gruHH   = (_Float16*)alloc((size_t)3 * HDIM * HDIM * 2);
  _Float16* lstmIH  = (_Float16*)alloc((size_t)4 * HDIM * 2 * HDIM * 2);   // Bt [512,256]
  _Float16* lstmHH  = (_Float16*)alloc((size_t)4 * HDIM * HDIM * 2);       // Bt [512,128]
  float*    hfeat   = (float*)alloc((size_t)NNODE * HDIM * 4);
  float*    update  = (float*)alloc((size_t)NNODE * HDIM * 4);
  _Float16* x16     = (_Float16*)alloc((size_t)NNODE * HDIM * 2);
  _Float16* h16     = (_Float16*)alloc((size_t)NNODE * HDIM * 2);
  float*    gi      = (float*)alloc((size_t)NNODE * 3 * HDIM * 4);
  float*    gh      = (float*)alloc((size_t)NNODE * 3 * HDIM * 4);
  float*    hS      = (float*)alloc((size_t)NB * HDIM * 4);
  float*    cS      = (float*)alloc((size_t)NB * HDIM * 4);
  _Float16* q16     = (_Float16*)alloc((size_t)NB * 2 * HDIM * 2);
  _Float16* h16s    = (_Float16*)alloc((size_t)NB * HDIM * 2);
  float*    g1      = (float*)alloc((size_t)NB * 4 * HDIM * 4);
  float*    g2      = (float*)alloc((size_t)NB * 4 * HDIM * 4);
  int*      startb  = (int*)alloc((size_t)(NB + 1) * 4);

  const int T256 = 256;
  // each wave covers a 16x64 strip -> tiles = (M/16)*(N/64), 8 waves per block
  auto gemm_blocks = [](long M, long N) { return cdiv((M >> 4) * (N >> 6), 8); };

  // ---- one-time weight conversions to f16 (Bt = B^T stored [N,K]) ----
  cvt_f32_f16<<<cdiv(NEDGE * EDIM, T256), T256, 0, stream>>>(edge_feat, edge16, NEDGE * EDIM);
  trans_cvt<<<cdiv(EDIM * HDIM, T256), T256, 0, stream>>>(mlp_w1, W1t, EDIM, HDIM);
  trans_cvt<<<cdiv(HDIM * HDIM * HDIM, T256), T256, 0, stream>>>(mlp_w2, W2t, HDIM, HDIM * HDIM);
  cvt_f32_f16<<<cdiv(3 * HDIM * HDIM, T256), T256, 0, stream>>>(gru_w_ih, gruIH, 3 * HDIM * HDIM);
  cvt_f32_f16<<<cdiv(3 * HDIM * HDIM, T256), T256, 0, stream>>>(gru_w_hh, gruHH, 3 * HDIM * HDIM);
  cvt_f32_f16<<<cdiv(4 * HDIM * 2 * HDIM, T256), T256, 0, stream>>>(lstm_w_ih, lstmIH, 4 * HDIM * 2 * HDIM);
  cvt_f32_f16<<<cdiv(4 * HDIM * HDIM, T256), T256, 0, stream>>>(lstm_w_hh, lstmHH, 4 * HDIM * HDIM);

  // ---- edge MLP stage 1: hmid = relu(edge @ W1 + b1)   [E,32]x[32,128] ----
  wmma_gemm<true, true><<<gemm_blocks(NEDGE, HDIM), T256, 0, stream>>>(
      edge16, W1t, mlp_b1, hmid16, NEDGE, HDIM, EDIM);

  // ---- node init: hfeat = embedding[unit_type] ----
  gather_emb<<<cdiv(NNODE * HDIM, T256), T256, 0, stream>>>(
      unit_type, embedding, hfeat, NNODE * HDIM);

  // ---- MPNN layers; transform recomputed in L2-resident 64 MiB chunks ----
  for (int layer = 0; layer < NLAYER; ++layer) {
    zero_f32<<<cdiv(NNODE * HDIM, T256), T256, 0, stream>>>(update, NNODE * HDIM);
    for (int e0 = 0; e0 < NEDGE; e0 += ECHUNK) {
      // Tchunk = hmid[e0:e0+CH] @ W2 + b2   [CH,128]x[128,16384] f16 out
      wmma_gemm<true, false><<<gemm_blocks(ECHUNK, HDIM * HDIM), T256, 0, stream>>>(
          hmid16 + (size_t)e0 * HDIM, W2t, mlp_b2, Tchunk, ECHUNK, HDIM * HDIM, HDIM);
      // msg + scatter-add, reading the chunk straight back out of L2
      edge_msg<<<ECHUNK, HDIM, 0, stream>>>(Tchunk, hfeat, node_in, node_out, update, e0);
    }
    relu_cvt<<<cdiv(NNODE * HDIM, T256), T256, 0, stream>>>(
        update, hfeat, x16, h16, NNODE * HDIM);
    wmma_gemm<false, false><<<gemm_blocks(NNODE, 3 * HDIM), T256, 0, stream>>>(
        x16, gruIH, gru_b_ih, gi, NNODE, 3 * HDIM, HDIM);
    wmma_gemm<false, false><<<gemm_blocks(NNODE, 3 * HDIM), T256, 0, stream>>>(
        h16, gruHH, gru_b_hh, gh, NNODE, 3 * HDIM, HDIM);
    gru_elem<<<cdiv(NNODE * HDIM, T256), T256, 0, stream>>>(gi, gh, hfeat, NNODE * HDIM);
  }

  // ---- Set2Set readout ----
  seg_starts<<<cdiv(NNODE, T256), T256, 0, stream>>>(node2graph, startb, NNODE, NB);
  zero_f32<<<cdiv(NB * HDIM, T256), T256, 0, stream>>>(hS, NB * HDIM);
  zero_f32<<<cdiv(NB * HDIM, T256), T256, 0, stream>>>(cS, NB * HDIM);
  float* qstar = out;  // first 512*256 floats of d_out hold q_star
  zero_f32<<<cdiv(NB * 2 * HDIM, T256), T256, 0, stream>>>(qstar, NB * 2 * HDIM);

  for (int s = 0; s < NS2S; ++s) {
    cvt_f32_f16<<<cdiv(NB * 2 * HDIM, T256), T256, 0, stream>>>(qstar, q16, NB * 2 * HDIM);
    cvt_f32_f16<<<cdiv(NB * HDIM, T256), T256, 0, stream>>>(hS, h16s, NB * HDIM);
    wmma_gemm<false, false><<<gemm_blocks(NB, 4 * HDIM), T256, 0, stream>>>(
        q16, lstmIH, lstm_b_ih, g1, NB, 4 * HDIM, 2 * HDIM);
    wmma_gemm<false, false><<<gemm_blocks(NB, 4 * HDIM), T256, 0, stream>>>(
        h16s, lstmHH, lstm_b_hh, g2, NB, 4 * HDIM, HDIM);
    lstm_elem<<<cdiv(NB * HDIM, T256), T256, 0, stream>>>(g1, g2, hS, cS, NB * HDIM);
    s2s_attn<<<NB, HDIM, 0, stream>>>(hfeat, hS, startb, qstar);
  }

  // ---- node_feature output ----
  copy_f32<<<cdiv(NNODE * HDIM, T256), T256, 0, stream>>>(
      hfeat, out + (size_t)NB * 2 * HDIM, NNODE * HDIM);
}
